// GaussianFeatureLeverage_67585605370671
// MI455X (gfx1250) — matrix operand order repaired and checked by reading, not compiled, not run on hardware
//
#include <hip/hip_runtime.h>

// ---------------------------------------------------------------------------
// GaussianFeatureLeverage for MI455X (gfx1250).
// Conv3x3 = implicit GEMM on v_wmma_f32_16x16x32_f16 (wave32), tap-major K,
// channel-last f16 activations so A and B fragments are both 2x b128 loads.
// ---------------------------------------------------------------------------

#define HH    96
#define WWID  320
#define NPIX  30720        // 96*320
#define BATCH 2
#define CFEAT 127

typedef __attribute__((ext_vector_type(16))) _Float16 v16h;
typedef __attribute__((ext_vector_type(8)))  _Float16 v8h;
typedef __attribute__((ext_vector_type(8)))  float    v8f;

__device__ __forceinline__ float sigmoid_f(float x) { return 1.0f / (1.0f + expf(-x)); }

template<int ACT>
__device__ __forceinline__ float activate(float v) {
  if (ACT == 1) return 0.5f * v * (1.0f + erff(v * 0.70710678118654752f));  // exact GELU
  if (ACT == 2) return v > 0.0f ? v : (expf(v) - 1.0f);                     // ELU
  return v;
}

// ------------------------- feature pyramid build ---------------------------
// jax.image.resize bilinear (half-pixel centers, edge clamp)
__device__ float bilin_up(const float* __restrict__ src, int sh, int sw, int y, int x) {
  float sy = (y + 0.5f) * ((float)sh / (float)HH) - 0.5f;
  float sx = (x + 0.5f) * ((float)sw / (float)WWID) - 0.5f;
  sy = fminf(fmaxf(sy, 0.0f), (float)(sh - 1));
  sx = fminf(fmaxf(sx, 0.0f), (float)(sw - 1));
  int y0 = (int)floorf(sy), x0 = (int)floorf(sx);
  int y1 = y0 + 1 < sh ? y0 + 1 : sh - 1;
  int x1 = x0 + 1 < sw ? x0 + 1 : sw - 1;
  float fy = sy - (float)y0, fx = sx - (float)x0;
  float v00 = src[y0 * sw + x0], v01 = src[y0 * sw + x1];
  float v10 = src[y1 * sw + x0], v11 = src[y1 * sw + x1];
  return v00 * (1.f - fy) * (1.f - fx) + v01 * (1.f - fy) * fx
       + v10 * fy * (1.f - fx) + v11 * fy * fx;
}

// feat[2][127][96][320] = concat(resized feats, resized disps, color1)
__global__ void build_feat_kernel(const float* __restrict__ f0, const float* __restrict__ d0,
                                  const float* __restrict__ f1, const float* __restrict__ d1,
                                  const float* __restrict__ f2, const float* __restrict__ d2,
                                  const float* __restrict__ f3, const float* __restrict__ d3,
                                  const float* __restrict__ c1, float* __restrict__ feat) {
  int idx = blockIdx.x * blockDim.x + threadIdx.x;
  const int total = BATCH * CFEAT * NPIX;
  if (idx >= total) return;
  int x = idx % WWID;
  int y = (idx / WWID) % HH;
  int c = (idx / NPIX) % CFEAT;
  int b = idx / (NPIX * CFEAT);
  float v;
  if (c < 16)        v = bilin_up(f0 + ((size_t)b * 16 + c)        * (48 * 160), 48, 160, y, x);
  else if (c < 40)   v = bilin_up(f1 + ((size_t)b * 24 + (c - 16)) * (24 * 80),  24, 80,  y, x);
  else if (c < 72)   v = bilin_up(f2 + ((size_t)b * 32 + (c - 40)) * (12 * 40),  12, 40,  y, x);
  else if (c < 120)  v = bilin_up(f3 + ((size_t)b * 48 + (c - 72)) * (6 * 20),   6,  20,  y, x);
  else if (c == 120) v = bilin_up(d0 + (size_t)b * (48 * 160), 48, 160, y, x);
  else if (c == 121) v = bilin_up(d1 + (size_t)b * (24 * 80),  24, 80,  y, x);
  else if (c == 122) v = bilin_up(d2 + (size_t)b * (12 * 40),  12, 40,  y, x);
  else if (c == 123) v = bilin_up(d3 + (size_t)b * (6 * 20),   6,  20,  y, x);
  else               v = c1[((size_t)b * 3 + (c - 124)) * NPIX + y * WWID + x];
  feat[idx] = v;
}

// --------------------------- SE-block pieces -------------------------------

__global__ void reduce_mean_kernel(const float* __restrict__ src, float* __restrict__ out, int hw) {
  __shared__ float sbuf[256];
  const int bc = blockIdx.x;
  const float* p = src + (size_t)bc * hw;
  float s = 0.0f;
  for (int i = threadIdx.x; i < hw; i += 256) s += p[i];
  sbuf[threadIdx.x] = s;
  __syncthreads();
  for (int off = 128; off > 0; off >>= 1) {
    if ((int)threadIdx.x < off) sbuf[threadIdx.x] += sbuf[threadIdx.x + off];
    __syncthreads();
  }
  if (threadIdx.x == 0) out[bc] = sbuf[0] / (float)hw;
}

__global__ void se_fc_kernel(const float* __restrict__ mean,
                             const float* __restrict__ w1, const float* __restrict__ b1,
                             const float* __restrict__ w2, const float* __restrict__ b2,
                             float* __restrict__ scale, int C, int r) {
  __shared__ float sm[160];
  __shared__ float hid[32];
  const int b = blockIdx.x;
  for (int c = threadIdx.x; c < C; c += blockDim.x) sm[c] = mean[b * C + c];
  __syncthreads();
  if ((int)threadIdx.x < r) {
    float s = b1[threadIdx.x];
    for (int c = 0; c < C; ++c) s += w1[threadIdx.x * C + c] * sm[c];
    hid[threadIdx.x] = fmaxf(s, 0.0f);
  }
  __syncthreads();
  for (int c = threadIdx.x; c < C; c += blockDim.x) {
    float s = b2[c];
    for (int j = 0; j < r; ++j) s += w2[c * r + j] * hid[j];
    scale[b * C + c] = sigmoid_f(s);
  }
}

// NCHW f32 -> NHWC f16 (channel padded with zeros), optional per-channel SE scale
__global__ void nchw_to_nhwc_f16_kernel(const float* __restrict__ src,
                                        const float* __restrict__ scale,
                                        _Float16* __restrict__ dst,
                                        int C, int CPad, int hw) {
  int idx = blockIdx.x * blockDim.x + threadIdx.x;
  const int total = BATCH * hw * CPad;
  if (idx >= total) return;
  const int c = idx % CPad;
  const int n = (idx / CPad) % hw;
  const int b = idx / (CPad * hw);
  float v = 0.0f;
  if (c < C) {
    v = src[((size_t)b * C + c) * hw + n];
    if (scale) v *= scale[b * C + c];
  }
  dst[idx] = (_Float16)v;
}

// -------------------- implicit-GEMM conv3x3 via WMMA -----------------------
// Activations: f16 NHWC [b][hw][CINPAD]. Weights: f16 [CoutPad][9*CINPAD],
// K index = t*CINPAD + cin (tap-major). One wave/block -> 16 couts x 16 pixels.
// OUT16: store f16 NHWC [b][hw][OUTPAD] (pad channels zeroed); else f32 NCHW.
template<int CIN, int COUT, int HD, int WD, int MODE, int ACT, bool OUT16>
__global__ __launch_bounds__(32)
void conv3x3_wmma_kernel(const _Float16* __restrict__ in16,
                         const _Float16* __restrict__ wpk,
                         const float* __restrict__ bias, void* __restrict__ outv) {
  constexpr int CINPAD = (CIN + 31) & ~31;
  constexpr int KPAD = 9 * CINPAD;
  constexpr int OUTPAD = (COUT + 31) & ~31;
  constexpr int HW = HD * WD;
  const int lane = threadIdx.x & 31;
  const int half = lane >> 4;
  const int l15  = lane & 15;
  const int ntile = blockIdx.x << 4;
  const int mtile = blockIdx.y << 4;
  const int bb    = blockIdx.z;

  const int n = ntile + l15;
  const bool nvalid = (n < HW);
  const int pyy = nvalid ? (n / WD) : 0;
  const int pxx = nvalid ? (n - pyy * WD) : 0;

  // A frag: lane row m=l15; elems 0..7 -> klocal=e+8*half, 8..15 -> 16+(e-8)+8*half
  const _Float16* __restrict__ wrow =
      wpk + (size_t)(mtile + l15) * KPAD + (half << 3);
  // B frag: lane col n; elems 0..7 -> klocal=e+8*half, 8..15 -> 16+(e-8)+8*half
  const _Float16* __restrict__ bbase =
      in16 + (size_t)bb * HW * CINPAD + (half << 3);

  v8f acc = {};
#pragma unroll
  for (int t = 0; t < 9; ++t) {
    const int dy = t / 3 - 1, dx = t % 3 - 1;
    int yy = pyy + dy, xx = pxx + dx;
    if (MODE == 0) {  // replicate
      yy = yy < 0 ? 0 : (yy >= HD ? HD - 1 : yy);
      xx = xx < 0 ? 0 : (xx >= WD ? WD - 1 : xx);
    } else {          // reflect
      yy = yy < 0 ? -yy : (yy >= HD ? 2 * HD - 2 - yy : yy);
      xx = xx < 0 ? -xx : (xx >= WD ? 2 * WD - 2 - xx : xx);
    }
    const _Float16* __restrict__ brow = bbase + (size_t)(yy * WD + xx) * CINPAD;
    const _Float16* __restrict__ wt = wrow + t * CINPAD;
#pragma unroll
    for (int cb = 0; cb < CINPAD; cb += 32) {
      const v8h alo = *(const v8h*)(wt + cb);
      const v8h ahi = *(const v8h*)(wt + cb + 16);
      const v8h blo = *(const v8h*)(brow + cb);
      const v8h bhi = *(const v8h*)(brow + cb + 16);
      v16h a, bm;
#pragma unroll
      for (int j = 0; j < 8; ++j) {
        a[j] = alo[j]; a[j + 8] = ahi[j];
        bm[j] = blo[j]; bm[j + 8] = bhi[j];
      }
      acc = __builtin_amdgcn_wmma_f32_16x16x32_f16(false, a, false, bm, (short)0, acc,
                                                   false, false);
    }
  }

  // C/D frag: lane l -> col n = l15; VGPR r -> row m = r + 8*half
  if (OUT16) {
    _Float16* out16 = (_Float16*)outv;
    if (nvalid) {
      v8h ov;
#pragma unroll
      for (int r = 0; r < 8; ++r) {
        const int m = mtile + r + (half << 3);
        float v = 0.0f;
        if (m < COUT) v = activate<ACT>(acc[r] + bias[m]);
        ov[r] = (_Float16)v;
      }
      *(v8h*)(out16 + ((size_t)bb * HW + n) * OUTPAD + mtile + (half << 3)) = ov;
    }
  } else {
    float* out = (float*)outv;
#pragma unroll
    for (int r = 0; r < 8; ++r) {
      const int m = mtile + r + (half << 3);
      if (nvalid && m < COUT) {
        out[((size_t)bb * COUT + m) * HW + n] = activate<ACT>(acc[r] + bias[m]);
      }
    }
  }
}

// Pack conv weight [O][I][3][3] f32 -> f16 [CoutPad][9*CinPad], k = t*CinPad+cin
__global__ void pack_weight_kernel(const float* __restrict__ w, _Float16* __restrict__ wpk,
                                   int Cout, int Cin, int CoutPad, int CinPad) {
  const int KP = 9 * CinPad;
  int idx = blockIdx.x * blockDim.x + threadIdx.x;
  const int total = CoutPad * KP;
  if (idx >= total) return;
  const int m = idx / KP;
  const int k = idx - m * KP;
  const int t = k / CinPad;
  const int cin = k - t * CinPad;
  float v = 0.0f;
  if (m < Cout && cin < Cin) v = w[((size_t)m * Cin + cin) * 9 + t];
  wpk[idx] = (_Float16)v;
}

// ------------------- gaussian parameters + backprojection ------------------
__global__ void gaussians_kernel(const float* __restrict__ feat, const float* __restrict__ h2,
                                 const float* __restrict__ invK, float* __restrict__ pos,
                                 float* __restrict__ smean, float* __restrict__ opac) {
  int idx = blockIdx.x * blockDim.x + threadIdx.x;
  if (idx >= BATCH * NPIX) return;
  const int b = idx / NPIX;
  const int n = idx - b * NPIX;
  const float* iK = invK + 9;  // invK[GS=1]
  const float disp = feat[((size_t)b * CFEAT + 120) * NPIX + n];
  float depth = 1.0f / (0.01f + 9.99f * disp);
  depth = fminf(fmaxf(depth, 0.1f), 100.0f);
  const float fx = (float)(n % WWID), fy = (float)(n / WWID);
  const float camx = iK[0] * fx + iK[1] * fy + iK[2];
  const float camy = iK[3] * fx + iK[4] * fy + iK[5];
  const float camz = iK[6] * fx + iK[7] * fy + iK[8];
  const float* g = h2 + (size_t)b * 11 * NPIX + n;
  pos[(size_t)idx * 3 + 0] = depth * camx + g[0 * NPIX];
  pos[(size_t)idx * 3 + 1] = depth * camy + g[1 * NPIX];
  pos[(size_t)idx * 3 + 2] = depth * camz + g[2 * NPIX];
  opac[idx]  = sigmoid_f(g[3 * NPIX]);
  smean[idx] = (expf(g[4 * NPIX]) + expf(g[5 * NPIX]) + expf(g[6 * NPIX])) * (0.01f / 3.0f);
}

__global__ void zero_f32_kernel(float* __restrict__ p, int n) {
  int i = blockIdx.x * blockDim.x + threadIdx.x;
  if (i < n) p[i] = 0.0f;
}

// ------------------------------ splatting ----------------------------------
__global__ void splat_kernel(const float* __restrict__ pos, const float* __restrict__ smean,
                             const float* __restrict__ opac, const float* __restrict__ gfeat,
                             const float* __restrict__ Ki, int hi, int wi,
                             float* __restrict__ num, float* __restrict__ den) {
  int idx = blockIdx.x * blockDim.x + threadIdx.x;
  if (idx >= BATCH * NPIX) return;
  const int b = idx / NPIX;
  const int n = idx - b * NPIX;
  const float px = pos[(size_t)idx * 3 + 0];
  const float py = pos[(size_t)idx * 3 + 1];
  const float pz = pos[(size_t)idx * 3 + 2];
  const float p0 = Ki[0] * px + Ki[1] * py + Ki[2] * pz;
  const float p1 = Ki[3] * px + Ki[4] * py + Ki[5] * pz;
  const float p2 = Ki[6] * px + Ki[7] * py + Ki[8] * pz;
  const float z = fmaxf(p2, 1e-6f);
  const float u = p0 / z, v = p1 / z;
  const int ix = (int)floorf(u), iy = (int)floorf(v);
  const bool valid = (pz > 0.1f) && (pz < 100.0f) && ix >= 0 && ix < wi && iy >= 0 && iy < hi;
  if (!valid) return;
  const float du = u - ((float)ix + 0.5f);
  const float dv = v - ((float)iy + 0.5f);
  const float sig = Ki[0] * smean[idx] / z;
  const float w = opac[idx] * expf(-(du * du + dv * dv) / (2.0f * sig * sig + 1e-6f));
  const int pix = iy * wi + ix;
  __hip_atomic_fetch_add(den + (size_t)b * hi * wi + pix, w,
                         __ATOMIC_RELAXED, __HIP_MEMORY_SCOPE_AGENT);
  float* np = num + ((size_t)b * hi * wi + pix) * 64;
  const float* gf = gfeat + (size_t)b * 64 * NPIX + n;
#pragma unroll 4
  for (int c = 0; c < 64; ++c)
    __hip_atomic_fetch_add(np + c, gf[(size_t)c * NPIX] * w,
                           __ATOMIC_RELAXED, __HIP_MEMORY_SCOPE_AGENT);
}

// fused = concat(lev(64), feats[i], colors[i+1], disps[i]) per level (f32 NCHW)
__global__ void fused_build_kernel(const float* __restrict__ num, const float* __restrict__ den,
                                   const float* __restrict__ fin, const float* __restrict__ col,
                                   const float* __restrict__ dsp, float* __restrict__ fused,
                                   int nch, int hi, int wi) {
  const int ci = 64 + nch + 4;
  const int hw = hi * wi;
  int idx = blockIdx.x * blockDim.x + threadIdx.x;
  const int total = BATCH * ci * hw;
  if (idx >= total) return;
  const int n = idx % hw;
  const int c = (idx / hw) % ci;
  const int b = idx / (hw * ci);
  float v;
  if (c < 64)
    v = num[((size_t)b * hw + n) * 64 + c] / (den[(size_t)b * hw + n] + 1e-6f);
  else if (c < 64 + nch)
    v = fin[((size_t)b * nch + (c - 64)) * hw + n];
  else if (c < 64 + nch + 3)
    v = col[((size_t)b * 3 + (c - 64 - nch)) * hw + n];
  else
    v = dsp[(size_t)b * hw + n];
  fused[idx] = v;
}

// ------------------------- per-level host helper ---------------------------
template<int CIc, int RIc, int NCHOc, int NCHIc, int LHc, int LWc>
static void run_level(const float* pos, const float* smean, const float* opac,
                      const float* gfeat, const float* Ki,
                      const float* fin, const float* col, const float* dsp,
                      const float* const* Pl, float* num, float* den,
                      float* fusedf, _Float16* fused16, const _Float16* wpk,
                      float* meanb, float* scaleb, float* outp, hipStream_t stream) {
  constexpr int hw = LHc * LWc;
  constexpr int CINPAD = (CIc + 31) & ~31;
  constexpr int GY = (NCHOc + 15) / 16;
  zero_f32_kernel<<<(BATCH * hw * 64 + 255) / 256, 256, 0, stream>>>(num, BATCH * hw * 64);
  zero_f32_kernel<<<(BATCH * hw + 255) / 256, 256, 0, stream>>>(den, BATCH * hw);
  splat_kernel<<<(BATCH * NPIX + 255) / 256, 256, 0, stream>>>(pos, smean, opac, gfeat,
                                                               Ki, LHc, LWc, num, den);
  fused_build_kernel<<<(BATCH * CIc * hw + 255) / 256, 256, 0, stream>>>(
      num, den, fin, col, dsp, fusedf, NCHIc, LHc, LWc);
  reduce_mean_kernel<<<BATCH * CIc, 256, 0, stream>>>(fusedf, meanb, hw);
  se_fc_kernel<<<BATCH, 128, 0, stream>>>(meanb, Pl[0], Pl[1], Pl[2], Pl[3], scaleb, CIc, RIc);
  nchw_to_nhwc_f16_kernel<<<(BATCH * hw * CINPAD + 255) / 256, 256, 0, stream>>>(
      fusedf, scaleb, fused16, CIc, CINPAD, hw);
  conv3x3_wmma_kernel<CIc, NCHOc, LHc, LWc, 1, 2, false>
      <<<dim3((hw + 15) / 16, GY, BATCH), 32, 0, stream>>>(fused16, wpk, Pl[5], outp);
}

// ------------------------------- launcher ----------------------------------

extern "C" void kernel_launch(void* const* d_in, const int* in_sizes, int n_in,
                              void* d_out, int out_size, void* d_ws, size_t ws_size,
                              hipStream_t stream) {
  (void)in_sizes; (void)n_in; (void)out_size; (void)ws_size;

  const float* feat_in[4] = {(const float*)d_in[0], (const float*)d_in[2],
                             (const float*)d_in[4], (const float*)d_in[6]};
  const float* disp_in[4] = {(const float*)d_in[1], (const float*)d_in[3],
                             (const float*)d_in[5], (const float*)d_in[7]};
  const float* color_in[5] = {(const float*)d_in[8], (const float*)d_in[9],
                              (const float*)d_in[10], (const float*)d_in[11],
                              (const float*)d_in[12]};
  const float* Kmat = (const float*)d_in[13];
  const float* invK = (const float*)d_in[14];
  const float* P[36];
  for (int i = 0; i < 36; ++i) P[i] = (const float*)d_in[15 + i];

  // ---- workspace layout ----
  char* ws = (char*)d_ws;
  size_t off = 0;
  auto alloc = [&](size_t bytes) -> void* {
    void* p = (void*)(ws + off);
    off += (bytes + 255) & ~(size_t)255;
    return p;
  };
  float*    feat   = (float*)alloc(sizeof(float) * BATCH * CFEAT * NPIX);
  _Float16* feat16 = (_Float16*)alloc(sizeof(_Float16) * BATCH * NPIX * 128);
  _Float16* x16    = (_Float16*)alloc(sizeof(_Float16) * BATCH * NPIX * 128); // SE(x), later l1
  _Float16* h1_16  = (_Float16*)alloc(sizeof(_Float16) * BATCH * NPIX * 32);  // later fused16
  float*    h2o    = (float*)alloc(sizeof(float) * BATCH * 11 * NPIX);
  float*    gfeat  = (float*)alloc(sizeof(float) * BATCH * 64 * NPIX);
  float*    pos    = (float*)alloc(sizeof(float) * BATCH * NPIX * 3);
  float*    smean  = (float*)alloc(sizeof(float) * BATCH * NPIX);
  float*    opac   = (float*)alloc(sizeof(float) * BATCH * NPIX);
  float*    num    = (float*)alloc(sizeof(float) * BATCH * 48 * 160 * 64);
  float*    den    = (float*)alloc(sizeof(float) * BATCH * 48 * 160);
  float*    fusedf = (float*)alloc(sizeof(float) * BATCH * 84 * 48 * 160);
  float*    meanb  = (float*)alloc(sizeof(float) * BATCH * 128);
  float*    scaleb = (float*)alloc(sizeof(float) * BATCH * 128);
  _Float16* wp_hc1 = (_Float16*)alloc((size_t)2 * 32 * 1152);
  _Float16* wp_hc2 = (_Float16*)alloc((size_t)2 * 16 * 288);
  _Float16* wp_lc1 = (_Float16*)alloc((size_t)2 * 128 * 1152);
  _Float16* wp_lc2 = (_Float16*)alloc((size_t)2 * 64 * 1152);
  _Float16* wp_r0  = (_Float16*)alloc((size_t)2 * 32 * 864);
  _Float16* wp_r1  = (_Float16*)alloc((size_t)2 * 48 * 864);
  _Float16* wp_r2  = (_Float16*)alloc((size_t)2 * 80 * 1152);
  _Float16* wp_r3  = (_Float16*)alloc((size_t)2 * 144 * 1152);

  auto packw = [&](const float* w, _Float16* dst, int Cout, int Cin, int CoutPad, int CinPad) {
    int total = CoutPad * 9 * CinPad;
    pack_weight_kernel<<<(total + 255) / 256, 256, 0, stream>>>(w, dst, Cout, Cin,
                                                                CoutPad, CinPad);
  };

  // ---- pack all conv weights to f16 tap-major GEMM layout ----
  packw(P[4],  wp_hc1, 22, 127, 32, 128);
  packw(P[6],  wp_hc2, 11, 22, 16, 32);
  packw(P[8],  wp_lc1, 127, 127, 128, 128);
  packw(P[10], wp_lc2, 64, 127, 64, 128);
  packw(P[16], wp_r0, 18, 84, 32, 96);
  packw(P[22], wp_r1, 36, 92, 48, 96);
  packw(P[28], wp_r2, 72, 100, 80, 128);
  packw(P[34], wp_r3, 144, 116, 144, 128);

  // ---- build 127-channel feature tensor ----
  {
    int total = BATCH * CFEAT * NPIX;
    build_feat_kernel<<<(total + 255) / 256, 256, 0, stream>>>(
        feat_in[0], disp_in[0], feat_in[1], disp_in[1], feat_in[2], disp_in[2],
        feat_in[3], disp_in[3], color_in[0], feat);
  }
  // raw feat in f16 NHWC (for l_c1)
  nchw_to_nhwc_f16_kernel<<<(BATCH * NPIX * 128 + 255) / 256, 256, 0, stream>>>(
      feat, nullptr, feat16, CFEAT, 128, NPIX);

  // ---- head: SE -> conv+gelu -> conv ----
  reduce_mean_kernel<<<BATCH * CFEAT, 256, 0, stream>>>(feat, meanb, NPIX);
  se_fc_kernel<<<BATCH, 128, 0, stream>>>(meanb, P[0], P[1], P[2], P[3], scaleb, CFEAT, 7);
  nchw_to_nhwc_f16_kernel<<<(BATCH * NPIX * 128 + 255) / 256, 256, 0, stream>>>(
      feat, scaleb, x16, CFEAT, 128, NPIX);  // x = feat * SE-scale, f16 NHWC
  conv3x3_wmma_kernel<127, 22, HH, WWID, 0, 1, true>
      <<<dim3(NPIX / 16, 2, BATCH), 32, 0, stream>>>(x16, wp_hc1, P[5], h1_16);
  conv3x3_wmma_kernel<22, 11, HH, WWID, 0, 0, false>
      <<<dim3(NPIX / 16, 1, BATCH), 32, 0, stream>>>(h1_16, wp_hc2, P[7], h2o);

  // ---- gaussian params (pos / smean / opac) ----
  gaussians_kernel<<<(BATCH * NPIX + 255) / 256, 256, 0, stream>>>(feat, h2o, invK,
                                                                   pos, smean, opac);

  // ---- per-gaussian features: elu(conv) x2 (reflect pad), raw feat input ----
  conv3x3_wmma_kernel<127, 127, HH, WWID, 1, 2, true>
      <<<dim3(NPIX / 16, 8, BATCH), 32, 0, stream>>>(feat16, wp_lc1, P[9], x16);
  conv3x3_wmma_kernel<127, 64, HH, WWID, 1, 2, false>
      <<<dim3(NPIX / 16, 4, BATCH), 32, 0, stream>>>(x16, wp_lc2, P[11], gfeat);

  // ---- per-level: splat -> normalize+concat -> SE -> elu(conv) ----
  _Float16* fused16 = h1_16;  // reuse
  float* outp = (float*)d_out;
  run_level<84, 5, 18, 16, 48, 160>(pos, smean, opac, gfeat, Kmat + 2 * 9,
      feat_in[0], color_in[1], disp_in[0], &P[12], num, den, fusedf, fused16,
      wp_r0, meanb, scaleb, outp + 0, stream);
  run_level<92, 5, 36, 24, 24, 80>(pos, smean, opac, gfeat, Kmat + 3 * 9,
      feat_in[1], color_in[2], disp_in[1], &P[18], num, den, fusedf, fused16,
      wp_r1, meanb, scaleb, outp + 276480, stream);
  run_level<100, 6, 72, 32, 12, 40>(pos, smean, opac, gfeat, Kmat + 4 * 9,
      feat_in[2], color_in[3], disp_in[2], &P[24], num, den, fusedf, fused16,
      wp_r2, meanb, scaleb, outp + 414720, stream);
  run_level<116, 7, 144, 48, 6, 20>(pos, smean, opac, gfeat, Kmat + 5 * 9,
      feat_in[3], color_in[4], disp_in[3], &P[30], num, den, fusedf, fused16,
      wp_r3, meanb, scaleb, outp + 483840, stream);
}
